// RelativeAttention_41661182771528
// MI455X (gfx1250) — matrix operand order repaired and checked by reading, not compiled, not run on hardware
//
#include <hip/hip_runtime.h>
#include <hip/hip_bf16.h>

typedef __attribute__((ext_vector_type(16))) __bf16 v16bf;
typedef __attribute__((ext_vector_type(8)))  __bf16 v8bf;
typedef __attribute__((ext_vector_type(8)))  float  v8f;

union FragB { v16bf v; uint4 q[2]; };
union Pack8 { v8bf h; uint4 q; };

typedef __attribute__((address_space(3))) const char lds_cchar;

// Native packed f32->bf16 conversion (8 lanes -> 16B).
__device__ __forceinline__ uint4 cvt8(float4 a, float4 b) {
  v8f f = {a.x, a.y, a.z, a.w, b.x, b.y, b.z, b.w};
  Pack8 p; p.h = __builtin_convertvector(f, v8bf);
  return p.q;
}
__device__ __forceinline__ uint4 cvt8v(v8f f) {
  Pack8 p; p.h = __builtin_convertvector(f, v8bf);
  return p.q;
}

// 32-bit LDS byte address of a __shared__ location (addrspace(3) cast).
__device__ __forceinline__ unsigned lds_addr(const void* p) {
  return (unsigned)(unsigned long long)(lds_cchar*)p;
}

// ASYNCcnt-tracked direct global->LDS copy (no VGPR data path).
__device__ __forceinline__ void async_ld_b128(const void* lds_dst, const void* gsrc) {
  asm volatile("global_load_async_to_lds_b128 %0, %1, off"
               :: "v"(lds_addr(lds_dst)), "v"(gsrc) : "memory");
}
__device__ __forceinline__ void wait_async0() {
  asm volatile("s_wait_asynccnt 0x0" ::: "memory");
}
// Async loads retire in order: <=8 outstanding means the oldest tile landed.
__device__ __forceinline__ void wait_async8() {
  asm volatile("s_wait_asynccnt 0x8" ::: "memory");
}

// C[M,N] = A[M,K] * B[N,K]^T  (K contiguous in both operands; batched via blockIdx.z)
// SRCF32: A,B f32 in global, converted to bf16 while staging (BK=32, single buffer).
//         else bf16 sources staged via triple-buffered global_load_async_to_lds_b128
//         (BK=64, partial asynccnt waits, two-tile-deep pipeline).
// EPI: 0 = store bf16 row-major, 1 = store bf16 transposed (C[n][m]),
//      2 = store f32 with scale + LDS-staged clipped relative bias, 3 = f32 plain.
template<bool SRCF32, int EPI>
__global__ __launch_bounds__(256) void gemm_nt_kernel(
    const void* __restrict__ Ag, const void* __restrict__ Bg,
    void* __restrict__ Cg, const float* __restrict__ bias,
    int Kd, int lda, int ldb, int ldc,
    long long strideA, long long strideB, long long strideC,
    float scale, int ML)
{
  constexpr int BM = 128, BN = 128;
  constexpr int BK   = SRCF32 ? 32 : 64;
  constexpr int NBUF = SRCF32 ? 1 : 3;
  constexpr int PAD  = 8;                       // conflict-free b128 row strides
  __shared__ unsigned short As[NBUF][BM][BK + PAD];
  __shared__ unsigned short Bs[NBUF][BN][BK + PAD];

  const int tid  = threadIdx.x;
  const int z    = blockIdx.z;
  const int bm   = blockIdx.y * BM;
  const int bn   = blockIdx.x * BN;

  const int wave = tid >> 5, lane = tid & 31;
  const int half = lane >> 4, lr = lane & 15;
  const int m0   = (wave & 3) * 32;   // 4 waves along M
  const int n0   = (wave >> 2) * 64;  // 2 waves along N

  // Stage the 255-wide relative-bias window for this block into LDS (EPI==2).
  __shared__ float sh_bias[256];
  int d0 = 0;
  if constexpr (EPI == 2) {
    d0 = bn - bm + (ML - 1) - 127;
    int d = d0 + tid;
    d = d < 0 ? 0 : (d > 2 * ML - 2 ? 2 * ML - 2 : d);
    sh_bias[tid] = bias[d];          // visibility covered by first main-loop barrier
  }

  v8f acc[2][4];
#pragma unroll
  for (int i = 0; i < 2; ++i)
#pragma unroll
    for (int j = 0; j < 4; ++j)
      acc[i][j] = (v8f){0.f,0.f,0.f,0.f,0.f,0.f,0.f,0.f};

  // Fragment loads per CDNA5 16-bit WMMA VGPR layouts + 8 WMMAs for one K-step.
  auto compute_tile = [&](int buf, int ks) {
    FragB a[2], b[4];
#pragma unroll
    for (int i = 0; i < 2; ++i) {
      const unsigned short* r = &As[buf][m0 + 16 * i + lr][ks];
      a[i].q[0] = *(const uint4*)(r + half * 8);        // K 0-7  / 8-15
      a[i].q[1] = *(const uint4*)(r + 16 + half * 8);   // K 16-23 / 24-31
    }
#pragma unroll
    for (int j = 0; j < 4; ++j) {
      const unsigned short* r = &Bs[buf][n0 + 16 * j + lr][ks];
      b[j].q[0] = *(const uint4*)(r + half * 8);        // K 0-7  / 8-15
      b[j].q[1] = *(const uint4*)(r + 16 + half * 8);   // K 16-23 / 24-31
    }
#pragma unroll
    for (int i = 0; i < 2; ++i)
#pragma unroll
      for (int j = 0; j < 4; ++j)
        acc[i][j] = __builtin_amdgcn_wmma_f32_16x16x32_bf16(
            false, a[i].v, false, b[j].v, (short)0, acc[i][j], false, false);
  };

  if constexpr (SRCF32) {
    const float* Af = (const float*)Ag + (size_t)z * strideA;
    const float* Bf = (const float*)Bg + (size_t)z * strideB;
    for (int kt = 0; kt < Kd; kt += BK) {
#pragma unroll
      for (int q2 = 0; q2 < 2; ++q2) {
        const int s   = tid + q2 * 256;     // 512 slots of 8 elements
        const int row = s >> 2, seg = s & 3;
        const float* pa = Af + (size_t)(bm + row) * lda + kt + seg * 8;
        const float* pb = Bf + (size_t)(bn + row) * ldb + kt + seg * 8;
        float4 a0 = *(const float4*)pa;
        float4 a1 = *(const float4*)(pa + 4);
        float4 b0 = *(const float4*)pb;
        float4 b1 = *(const float4*)(pb + 4);
        *(uint4*)&As[0][row][seg * 8] = cvt8(a0, a1);
        *(uint4*)&Bs[0][row][seg * 8] = cvt8(b0, b1);
        if (kt + BK < Kd) {
          __builtin_prefetch(pa + BK, 0, 1);   // global_prefetch_b8 next K-tile
          __builtin_prefetch(pb + BK, 0, 1);
        }
      }
      __syncthreads();
      compute_tile(0, 0);
      __syncthreads();
    }
  } else {
    const unsigned short* Ab = (const unsigned short*)Ag + (size_t)z * strideA;
    const unsigned short* Bb = (const unsigned short*)Bg + (size_t)z * strideB;

    // One BK=64 tile: 8 async b128 per thread (4 for A, 4 for B).
    auto issue_tile = [&](int kt, int buf) {
#pragma unroll
      for (int q4 = 0; q4 < 4; ++q4) {
        const int s   = tid + q4 * 256;     // 1024 slots of 8 elements per matrix
        const int row = s >> 3, seg = s & 7;
        async_ld_b128(&As[buf][row][seg * 8],
                      Ab + (size_t)(bm + row) * lda + kt + seg * 8);
        async_ld_b128(&Bs[buf][row][seg * 8],
                      Bb + (size_t)(bn + row) * ldb + kt + seg * 8);
      }
    };

    issue_tile(0, 0);
    if (BK < Kd) issue_tile(BK, 1);
    int buf = 0;
    for (int kt = 0; kt < Kd; kt += BK) {
      if (kt + BK < Kd) wait_async8();   // oldest tile landed, next stays in flight
      else              wait_async0();   // last tile: drain fully
      __syncthreads();                   // all waves landed; buffer (buf+2)%3 free
      if (kt + 2 * BK < Kd) issue_tile(kt + 2 * BK, (buf + 2) % 3);
      compute_tile(buf, 0);              // 16 WMMAs overlap in-flight asyncs
      compute_tile(buf, 32);
      buf = (buf + 1) % 3;
    }
  }

  // ---- epilogue ----
  const int cmb = bm + m0, cnb = bn + n0;
  if constexpr (EPI == 0) {               // bf16 row-major (Q, K)
    __bf16* Cb = (__bf16*)Cg + (size_t)z * strideC;
#pragma unroll
    for (int i = 0; i < 2; ++i)
#pragma unroll
      for (int j = 0; j < 4; ++j) {
        const int gc = cnb + 16 * j + lr;
#pragma unroll
        for (int r = 0; r < 8; ++r) {
          const int gr = cmb + 16 * i + 8 * half + r;
          Cb[(size_t)gr * ldc + gc] = (__bf16)acc[i][j][r];
        }
      }
  } else if constexpr (EPI == 1) {        // bf16 transposed (V^T): contiguous b128
    unsigned short* Cb = (unsigned short*)Cg + (size_t)z * strideC;
#pragma unroll
    for (int i = 0; i < 2; ++i)
#pragma unroll
      for (int j = 0; j < 4; ++j) {
        const int gc = cnb + 16 * j + lr;        // output row (N dim)
        const int gr = cmb + 16 * i + 8 * half;  // output col base (M dim)
        *(uint4*)&Cb[(size_t)gc * ldc + gr] = cvt8v(acc[i][j]);
      }
  } else {                                // f32 (scores w/ bias, or plain out)
    float* Cf = (float*)Cg + (size_t)z * strideC;
#pragma unroll
    for (int i = 0; i < 2; ++i)
#pragma unroll
      for (int j = 0; j < 4; ++j) {
        const int gc = cnb + 16 * j + lr;
#pragma unroll
        for (int r = 0; r < 8; ++r) {
          const int gr = cmb + 16 * i + 8 * half + r;
          float v = acc[i][j][r];
          if constexpr (EPI == 2) {
            // window index: (gc-gr) - (bn-bm) + 127 in [0,254]
            v = v * scale + sh_bias[(gc - gr) + (ML - 1) - d0];
          }
          Cf[(size_t)gr * ldc + gc] = v;
        }
      }
  }
}

// Row softmax over 2048 elements; writes normalized f32 in-place and a bf16 copy.
__global__ __launch_bounds__(256) void softmax_rows_kernel(
    float* __restrict__ W, __bf16* __restrict__ P, int Cn)
{
  const int row = blockIdx.x;
  const int tid = threadIdx.x;
  float* rw = W + (size_t)row * Cn;
  __bf16* rp = P + (size_t)row * Cn;

  float v[8];
#pragma unroll
  for (int i = 0; i < 8; ++i) v[i] = rw[tid + i * 256];

  float mx = v[0];
#pragma unroll
  for (int i = 1; i < 8; ++i) mx = fmaxf(mx, v[i]);
#pragma unroll
  for (int off = 16; off > 0; off >>= 1) mx = fmaxf(mx, __shfl_xor(mx, off, 32));

  __shared__ float red[8];
  const int wave = tid >> 5, lane = tid & 31;
  if (lane == 0) red[wave] = mx;
  __syncthreads();
  float m_all = red[0];
#pragma unroll
  for (int w = 1; w < 8; ++w) m_all = fmaxf(m_all, red[w]);
  __syncthreads();

  float e[8], sum = 0.f;
#pragma unroll
  for (int i = 0; i < 8; ++i) { e[i] = __expf(v[i] - m_all); sum += e[i]; }
#pragma unroll
  for (int off = 16; off > 0; off >>= 1) sum += __shfl_xor(sum, off, 32);
  if (lane == 0) red[wave] = sum;
  __syncthreads();
  float s_all = 0.f;
#pragma unroll
  for (int w = 0; w < 8; ++w) s_all += red[w];
  const float inv = 1.0f / s_all;

#pragma unroll
  for (int i = 0; i < 8; ++i) {
    const float wv = e[i] * inv;
    rw[tid + i * 256] = wv;
    rp[tid + i * 256] = (__bf16)wv;
  }
}

extern "C" void kernel_launch(void* const* d_in, const int* in_sizes, int n_in,
                              void* d_out, int out_size, void* d_ws, size_t ws_size,
                              hipStream_t stream) {
  (void)in_sizes; (void)n_in; (void)out_size; (void)ws_size;
  constexpr int B = 4, C = 2048, E = 1024, ML = 2048;

  const float* x    = (const float*)d_in[0];
  const float* Wq   = (const float*)d_in[1];
  const float* Wk   = (const float*)d_in[2];
  const float* Wv   = (const float*)d_in[3];
  const float* bias = (const float*)d_in[4];
  // d_in[5] = max_len (known statically: 2048)

  float* out = (float*)d_out;                       // [B,C,E] f32
  float* wts = (float*)d_out + (size_t)B * C * E;   // [B,C,C] f32

  char* ws = (char*)d_ws;
  unsigned short* Qb = (unsigned short*)(ws);                              // [B*C, E] bf16
  unsigned short* Kb = (unsigned short*)(ws + (size_t)1 * B * C * E * 2);  // [B*C, E] bf16
  unsigned short* Vt = (unsigned short*)(ws + (size_t)2 * B * C * E * 2);  // [E, B*C] bf16
  unsigned short* Pb = (unsigned short*)(ws + (size_t)3 * B * C * E * 2);  // [B*C, C] bf16

  const float scl = 0.03125f;  // 1/sqrt(1024)

  // --- QKV projections: [8192,1024] x [1024,1024]^T (f32 -> bf16) ---
  dim3 gQKV(E / 128, (B * C) / 128, 1);
  gemm_nt_kernel<true, 0><<<gQKV, 256, 0, stream>>>(
      x, Wq, Qb, nullptr, E, E, E, E, 0, 0, 0, 1.f, ML);
  gemm_nt_kernel<true, 0><<<gQKV, 256, 0, stream>>>(
      x, Wk, Kb, nullptr, E, E, E, E, 0, 0, 0, 1.f, ML);
  gemm_nt_kernel<true, 1><<<gQKV, 256, 0, stream>>>(     // V stored transposed
      x, Wv, Vt, nullptr, E, E, E, B * C, 0, 0, 0, 1.f, ML);

  // --- scores = Q K^T / sqrt(E) + rel_bias  -> d_out weights region ---
  dim3 gS(C / 128, C / 128, B);
  gemm_nt_kernel<false, 2><<<gS, 256, 0, stream>>>(
      Qb, Kb, wts, bias, E, E, E, C,
      (long long)C * E, (long long)C * E, (long long)C * C, scl, ML);

  // --- softmax rows (in-place f32) + bf16 copy for PV GEMM ---
  softmax_rows_kernel<<<B * C, 256, 0, stream>>>(wts, (__bf16*)Pb, C);

  // --- out = weights @ V  (NT vs V^T) -> d_out out region ---
  dim3 gO(E / 128, C / 128, B);
  gemm_nt_kernel<false, 3><<<gO, 256, 0, stream>>>(
      Pb, Vt, out, nullptr, C, C, B * C, E,
      (long long)C * C, (long long)C, (long long)C * E, 1.f, ML);
}
